// GraphConvolution_LEGNN_53790170415695
// MI455X (gfx1250) — matrix operand order repaired and checked by reading, not compiled, not run on hardware
//
#include <hip/hip_runtime.h>

#define NN   10000
#define FF   256
#define CC   16
#define KPAD 10016
#define ALPHA 0.1f
#define EPSN  1e-12f

typedef __bf16 bf16t;
typedef __attribute__((ext_vector_type(16))) bf16t v16bf;
typedef __attribute__((ext_vector_type(8)))  float v8f;

union FragAB { v16bf v; unsigned short us[16]; unsigned int ui[8]; };
union F8     { float4 f4[2]; float f[8]; };

__device__ __forceinline__ float lane_bcast(float v, int src) {
  return __builtin_bit_cast(float,
      __builtin_amdgcn_ds_bpermute(src << 2, __builtin_bit_cast(int, v)));
}

__device__ __forceinline__ v8f vzero8() {
  v8f z = {0.f, 0.f, 0.f, 0.f, 0.f, 0.f, 0.f, 0.f};
  return z;
}

// ---- prep: labels from one-hot y -------------------------------------------
__global__ void lg_prep_labels(const float* __restrict__ y,
                               unsigned char* __restrict__ lab) {
  int i = blockIdx.x * blockDim.x + threadIdx.x;
  if (i >= KPAD) return;
  if (i >= NN) { lab[i] = 0xFF; return; }
  const float* yr = y + (size_t)i * CC;
  int best = 0; float bv = yr[0];
#pragma unroll
  for (int c = 1; c < CC; ++c) { float v = yr[c]; if (v > bv) { bv = v; best = c; } }
  lab[i] = (unsigned char)best;
}

// ---- prep: x (f32, row-major [N][F]) -> bf16 transposed [F][KPAD] ----------
__global__ void lg_prep_xbt(const float* __restrict__ x,
                            unsigned short* __restrict__ xbt) {
  int idx = blockIdx.x * blockDim.x + threadIdx.x;
  if (idx >= FF * KPAD) return;
  int kk  = idx >> 8;        // idx / 256
  int col = idx & 255;       // idx % 256
  unsigned short r = 0;
  if (kk < NN) {
    unsigned int u = __builtin_bit_cast(unsigned int, x[(size_t)kk * FF + col]);
    u += 0x7FFFu + ((u >> 16) & 1u);   // round-to-nearest-even f32->bf16
    r = (unsigned short)(u >> 16);
  }
  xbt[(size_t)col * KPAD + kk] = r;
}

// ---- fused masked-GEMM: out = 0.9 * rownorm(A*mask) @ x + 0.1 * h0 ---------
__global__ __launch_bounds__(32) void lg_gemm(
    const float* __restrict__ A, const float* __restrict__ h0,
    const unsigned short* __restrict__ xbt, const unsigned char* __restrict__ lab,
    float* __restrict__ out) {
  const int lane = threadIdx.x & 31;
  const int h    = lane >> 4;     // wave half (ISA A/B fragment layout)
  const int mr   = lane & 15;     // row-in-tile for A, col-in-tile for B
  const int tile0 = blockIdx.x * 2;

  v8f acc[2][16];
#pragma unroll
  for (int t = 0; t < 2; ++t)
#pragma unroll
    for (int cb = 0; cb < 16; ++cb) acc[t][cb] = vzero8();

  float rs[2] = {0.f, 0.f};               // per-lane partial row sums (L1 norm)
  bool rv[2]; unsigned char labi[2]; const float* arow[2];
#pragma unroll
  for (int t = 0; t < 2; ++t) {
    int r  = (tile0 + t) * 16 + mr;
    rv[t]  = r < NN;
    labi[t] = rv[t] ? lab[r] : (unsigned char)0xFE;
    arow[t] = A + (size_t)(rv[t] ? r : 0) * NN;
  }

  for (int k = 0; k < NN; k += 32) {
    const bool full = (k + 32) <= NN;     // last step has only 16 valid cols
    const int c0 = k + h * 8;             // A-frag K = h*8 + {0..7}
    const int c1 = c0 + 16;               // A-frag K = 16 + h*8 + {0..7}

    unsigned long long l0 = *(const unsigned long long*)(lab + c0);
    unsigned long long l1 = full ? *(const unsigned long long*)(lab + c1)
                                 : 0xFFFFFFFFFFFFFFFFull;
    FragAB af[2];
#pragma unroll
    for (int t = 0; t < 2; ++t) {
      F8 g0, g1;
      if (rv[t]) {
        __builtin_prefetch(arow[t] + c0 + 256, 0, 0);
        g0.f4[0] = *(const float4*)(arow[t] + c0);
        g0.f4[1] = *(const float4*)(arow[t] + c0 + 4);
      } else {
        g0.f4[0] = make_float4(0.f, 0.f, 0.f, 0.f);
        g0.f4[1] = make_float4(0.f, 0.f, 0.f, 0.f);
      }
      if (rv[t] && full) {
        g1.f4[0] = *(const float4*)(arow[t] + c1);
        g1.f4[1] = *(const float4*)(arow[t] + c1 + 4);
      } else {
        g1.f4[0] = make_float4(0.f, 0.f, 0.f, 0.f);
        g1.f4[1] = make_float4(0.f, 0.f, 0.f, 0.f);
      }
#pragma unroll
      for (int e = 0; e < 8; ++e) {
        unsigned char j0 = (unsigned char)(l0 >> (8 * e));
        unsigned char j1 = (unsigned char)(l1 >> (8 * e));
        float m0 = (j0 == labi[t]) ? g0.f[e] : 0.0f;   // A is 0/1 -> bf16 exact
        float m1 = (j1 == labi[t]) ? g1.f[e] : 0.0f;
        rs[t] += m0 + m1;
        af[t].us[e]     = (unsigned short)(__builtin_bit_cast(unsigned int, m0) >> 16);
        af[t].us[e + 8] = (unsigned short)(__builtin_bit_cast(unsigned int, m1) >> 16);
      }
    }

    const int kb = k + h * 8;             // B-frag K = h*8 + {0..7}, 16 + h*8 + {0..7}
#pragma unroll
    for (int cb = 0; cb < 16; ++cb) {
      const uint4* bp = (const uint4*)(xbt + (size_t)(cb * 16 + mr) * KPAD + kb);
      uint4 q0 = bp[0];
      uint4 q1 = bp[2];
      FragAB bfr;
      bfr.ui[0] = q0.x; bfr.ui[1] = q0.y; bfr.ui[2] = q0.z; bfr.ui[3] = q0.w;
      bfr.ui[4] = q1.x; bfr.ui[5] = q1.y; bfr.ui[6] = q1.z; bfr.ui[7] = q1.w;
      acc[0][cb] = __builtin_amdgcn_wmma_f32_16x16x32_bf16(
          false, af[0].v, false, bfr.v, (short)0, acc[0][cb], false, false);
      acc[1][cb] = __builtin_amdgcn_wmma_f32_16x16x32_bf16(
          false, af[1].v, false, bfr.v, (short)0, acc[1][cb], false, false);
    }
  }

  // Epilogue: cross-lane row-sum reduce, scale, blend with h0, store.
#pragma unroll
  for (int t = 0; t < 2; ++t) {
    float inv[8];
#pragma unroll
    for (int e = 0; e < 8; ++e) {
      int r = h * 8 + e;                  // C/D layout: VGPR e -> row e + 8*h
      float sum = lane_bcast(rs[t], r) + lane_bcast(rs[t], r + 16);
      inv[e] = (1.0f - ALPHA) / fmaxf(sum, EPSN);
    }
    int tb = (tile0 + t) * 16;
#pragma unroll
    for (int e = 0; e < 8; ++e) {
      int grow = tb + h * 8 + e;
      if (grow < NN) {
        float*       orow = out + (size_t)grow * FF + mr;
        const float* hrow = h0  + (size_t)grow * FF + mr;
#pragma unroll
        for (int cb = 0; cb < 16; ++cb)
          orow[cb * 16] = acc[t][cb][e] * inv[e] + ALPHA * hrow[cb * 16];
      }
    }
  }
}

extern "C" void kernel_launch(void* const* d_in, const int* in_sizes, int n_in,
                              void* d_out, int out_size, void* d_ws, size_t ws_size,
                              hipStream_t stream) {
  const float* x  = (const float*)d_in[0];   // "input" [N,F]
  const float* A  = (const float*)d_in[1];   // "A"     [N,N]
  const float* h0 = (const float*)d_in[2];   // "h0"    [N,F]
  const float* y  = (const float*)d_in[3];   // "y"     [N,C]
  float* out = (float*)d_out;

  unsigned short* xbt = (unsigned short*)d_ws;                       // 256*10016*2 B
  unsigned char*  lab = (unsigned char*)d_ws + (size_t)FF * KPAD * 2; // 10016 B

  lg_prep_labels<<<(KPAD + 255) / 256, 256, 0, stream>>>(y, lab);
  lg_prep_xbt<<<(FF * KPAD + 255) / 256, 256, 0, stream>>>(x, xbt);

  const int tiles  = (NN + 15) / 16;        // 625
  const int blocks = (tiles + 1) / 2;       // 313 waves, 2 row-tiles each
  lg_gemm<<<blocks, 32, 0, stream>>>(A, h0, xbt, lab, out);
}
